// TrajectoryDiscriminator_4217657885350
// MI455X (gfx1250) — matrix operand order repaired and checked by reading, not compiled, not run on hardware
//
#include <hip/hip_runtime.h>

#define BB 65536
#define TT 20

typedef __attribute__((ext_vector_type(16))) __bf16 v16bf;
typedef __attribute__((ext_vector_type(8)))  float  v8f;

union FragAB {
    unsigned u[8];
    uint4    q[2];
    v16bf    v;
};

// ---------- helpers ----------
__device__ inline unsigned f2bf1s(float f) {           // f32 -> bf16 bits (round w/ bias)
    return (__float_as_uint(f) + 0x8000u) >> 16;
}
__device__ inline unsigned packbf(float a, float b) {  // 2 adds + v_perm_b32
    unsigned ua = __float_as_uint(a) + 0x8000u;
    unsigned ub = __float_as_uint(b) + 0x8000u;
    return __builtin_amdgcn_perm(ub, ua, 0x07060302u); // lo16=ua[31:16], hi16=ub[31:16]
}
__device__ inline float bf2f(unsigned short s) {
    return __uint_as_float(((unsigned)s) << 16);
}
__device__ inline float sigmoidf_(float x) { return 1.0f / (1.0f + __expf(-x)); }
__device__ inline float tanhf_(float x) {   // stable: ±inf-safe
    float e = __expf(2.0f * x);
    return 1.0f - 2.0f / (e + 1.0f);
}
__device__ inline void ldsfence() {
    asm volatile("s_wait_dscnt 0" ::: "memory");
}
__device__ inline void schedbar() {
    __builtin_amdgcn_sched_barrier(0);     // keep B-frag prefetch window small
}
__device__ inline v8f wmma_bf16(v16bf a, v16bf b, v8f c) {
    return __builtin_amdgcn_wmma_f32_16x16x32_bf16(false, a, false, b, (short)0, c, false, false);
}

// B fragment: pre-swizzled in LDS as [frag][lane][8 dwords] (dword v packs K=k0,k0+1 with
// k0 = (v>>2)*16 + half*8 + 2*(v&3), matching the A layout; N = lane&15).
__device__ inline v16bf loadBfrag(const unsigned* swz, int frag, int lane) {
    FragAB f;
    const uint4* p = reinterpret_cast<const uint4*>(swz + frag * 256 + lane * 8);
    f.q[0] = p[0];
    f.q[1] = p[1];
    return f.v;
}

// A fragment from a row-major bf16 LDS tile: lane half p holds K = p*8..p*8+7 and
// 16+p*8..16+p*8+7 of row (lane&15), per the 16-bit A layout.
__device__ inline v16bf loadAfrag(const unsigned short* tile, int strideU16, int colBase, int lane) {
    const int p = lane >> 4, m = lane & 15;
    FragAB f;
    f.q[0] = *reinterpret_cast<const uint4*>(tile + m * strideU16 + colBase + p * 8);
    f.q[1] = *reinterpret_cast<const uint4*>(tile + m * strideU16 + colBase + 16 + p * 8);
    return f.v;
}

// Fill one swizzled B fragment from a row-major weight W[NOUT][KIN] used as B[k][n]=W[n][k].
__device__ inline void fillBfrag(unsigned* dst, const float* W, int KIN, int ktile, int ntile,
                                 int tid, int nthr) {
    for (int idx = tid; idx < 256; idx += nthr) {
        int l = idx >> 3, v = idx & 7;
        int n  = ntile * 16 + (l & 15);
        int k0 = ktile * 32 + ((v >> 2) * 16) + ((l >> 4) * 8) + 2 * (v & 3);
        float f0 = (k0     < KIN) ? W[n * KIN + k0]     : 0.0f;
        float f1 = (k0 + 1 < KIN) ? W[n * KIN + k0 + 1] : 0.0f;
        dst[idx] = packbf(f0, f1);
    }
}

// ---------- LDS layout (dynamic) ----------
// frags:  pWih 0..7 | pWhh 8..15 | mWih 16..47 [kt*16+nt] | mWhh 48..79 | W2pos 80 | W2light 81
#define RAWF_OFF   83968
#define WSCR_OFF   93504
#define WSCR_SIZE  5632
#define SMEM_BYTES 116032

// ---------- stats pipeline ----------
__global__ void zero_kernel(float* acc) {
    int i = blockIdx.x * blockDim.x + threadIdx.x;
    if (i < 512) acc[i] = 0.0f;
}

__global__ void stats_kernel(const float* __restrict__ traj, const float* __restrict__ state,
                             float* __restrict__ acc) {
    int t = blockIdx.x >> 4;
    int slice = blockIdx.x & 15;
    int b0 = slice * (BB / 16);
    __shared__ float red[12];
    if (threadIdx.x < 12) red[threadIdx.x] = 0.0f;
    __syncthreads();
    float s[12];
#pragma unroll
    for (int i = 0; i < 12; i++) s[i] = 0.0f;
    for (int b = b0 + threadIdx.x; b < b0 + BB / 16; b += 256) {
        const float* tr = traj + (size_t)t * BB * 2 + (size_t)b * 2;
        float x0 = tr[0], x1 = tr[1];
        s[0] += x0; s[1] += x0 * x0; s[2] += x1; s[3] += x1 * x1;
        const float* st = state + (size_t)t * BB * 4 + (size_t)b * 4;
#pragma unroll
        for (int f = 0; f < 4; f++) {
            float x = st[f];
            s[4 + 2 * f] += x;
            s[5 + 2 * f] += x * x;
        }
    }
#pragma unroll
    for (int i = 0; i < 12; i++) atomicAdd(&red[i], s[i]);
    __syncthreads();
    if (threadIdx.x < 12) {
        int i = threadIdx.x;
        int idx;
        if (i < 4) { int f = i >> 1; idx = (t * 2 + f) * 2 + (i & 1); }
        else       { int k = i - 4; int f = k >> 1; idx = 80 + (t * 4 + f) * 2 + (k & 1); }
        atomicAdd(&acc[idx], red[i]);
    }
}

__global__ void finalize_kernel(const float* __restrict__ acc,
                                const float* pg, const float* pb,
                                const float* lg, const float* lb,
                                float* __restrict__ ssp) {
    int i = threadIdx.x;
    if (i < 40) {                       // traj: (t,f) pairs
        int t = i >> 1, f = i & 1;
        float sum = acc[(t * 2 + f) * 2], sq = acc[(t * 2 + f) * 2 + 1];
        float mean = sum / (float)BB;
        float var  = sq / (float)BB - mean * mean;
        float inv  = rsqrtf(var + 1e-5f);
        float sc   = pg[f] * inv;
        ssp[(t * 2 + f) * 2]     = sc;
        ssp[(t * 2 + f) * 2 + 1] = pb[f] - mean * sc;
    } else if (i < 120) {               // state
        int k = i - 40;
        int t = k >> 2, f = k & 3;
        float sum = acc[80 + (t * 4 + f) * 2], sq = acc[80 + (t * 4 + f) * 2 + 1];
        float mean = sum / (float)BB;
        float var  = sq / (float)BB - mean * mean;
        float inv  = rsqrtf(var + 1e-5f);
        float sc   = lg[f] * inv;
        ssp[80 + (t * 4 + f) * 2]     = sc;
        ssp[80 + (t * 4 + f) * 2 + 1] = lb[f] - mean * sc;
    }
}

// ---------- per-stream LSTM step (S=0: pos/traj, S=1: light/state) ----------
// Gate tiles are computed per column-group j so only 4 accumulator fragments
// (i,f,g,o for that group) are live at once; sched barriers keep the LDS
// B-fragment prefetch window small so demand stays under 256 VGPRs.
template <int S>
__device__ inline void part_stream_step(int t, int row0, int lane,
                                        const float* __restrict__ xglob,
                                        const float* __restrict__ ss,
                                        const float* __restrict__ rawf,
                                        const unsigned* __restrict__ swz,
                                        unsigned short* eTile, unsigned short* mgTile, float* xbn,
                                        float b2v, const float (&pb)[8], v8f (&c)[2]) {
    const int p = lane >> 4, nn = lane & 15;
    constexpr int KIN = (S == 0) ? 2 : 4;

    if (lane < 16) {                    // BN'd inputs -> LDS (one row per lane)
        int m = lane;
#pragma unroll
        for (int f = 0; f < KIN; f++) {
            float sc = ss[(t * KIN + f) * 2];
            float sh = ss[(t * KIN + f) * 2 + 1];
            float x  = xglob[(size_t)t * BB * KIN + (size_t)(row0 + m) * KIN + f];
            xbn[m * 4 + f] = x * sc + sh;
        }
    }
    ldsfence();

    // MLP1 (KIN->32), computed straight into bf16 A-fragment register layout
    const float* W1 = (S == 0) ? rawf : (rawf + 64);
    const float* b1 = (S == 0) ? (rawf + 192) : (rawf + 224);
    FragAB aH;
    {
        int m = nn;
#pragma unroll
        for (int vi = 0; vi < 8; vi++) {
            int kA = (vi < 4) ? (p * 8 + 2 * vi) : (16 + p * 8 + 2 * (vi - 4));
            float a0 = b1[kA], a1 = b1[kA + 1];
#pragma unroll
            for (int f = 0; f < KIN; f++) {
                float x = xbn[m * 4 + f];
                a0 += x * W1[kA * KIN + f];
                a1 += x * W1[(kA + 1) * KIN + f];
            }
            aH.u[vi] = packbf(fmaxf(a0, 0.0f), fmaxf(a1, 0.0f));
        }
    }

    // MLP2 (32->16) via WMMA, bias in C, ReLU, write A-tile for the LSTM
    v16bf w2 = loadBfrag(swz, (S == 0) ? 80 : 81, lane);
    v8f e;
#pragma unroll
    for (int r = 0; r < 8; r++) e[r] = b2v;
    e = wmma_bf16(aH.v, w2, e);
    schedbar();
#pragma unroll
    for (int r = 0; r < 8; r++)
        eTile[(r + 8 * p) * 32 + nn] = (unsigned short)f2bf1s(fmaxf(e[r], 0.0f));
    ldsfence();

    // LSTM gates = e @ Wih^T + h_prev @ Whh^T + (bih+bhh); group by c-column tile j
    v16bf aE  = loadAfrag(eTile, 32, 0, lane);
    v16bf aHp = loadAfrag(mgTile, 64, S * 32, lane);
#pragma unroll
    for (int j = 0; j < 2; j++) {
        v8f gi, gf, gg, go;
#pragma unroll
        for (int r = 0; r < 8; r++) {
            gi[r] = pb[j]; gf[r] = pb[2 + j]; gg[r] = pb[4 + j]; go[r] = pb[6 + j];
        }
        gi = wmma_bf16(aE,  loadBfrag(swz, j,      lane), gi);
        gi = wmma_bf16(aHp, loadBfrag(swz, 8 + j,  lane), gi);
        schedbar();
        gf = wmma_bf16(aE,  loadBfrag(swz, 2 + j,  lane), gf);
        gf = wmma_bf16(aHp, loadBfrag(swz, 10 + j, lane), gf);
        schedbar();
        gg = wmma_bf16(aE,  loadBfrag(swz, 4 + j,  lane), gg);
        gg = wmma_bf16(aHp, loadBfrag(swz, 12 + j, lane), gg);
        schedbar();
        go = wmma_bf16(aE,  loadBfrag(swz, 6 + j,  lane), go);
        go = wmma_bf16(aHp, loadBfrag(swz, 14 + j, lane), go);
        schedbar();
#pragma unroll
        for (int r = 0; r < 8; r++) {
            float iv = sigmoidf_(gi[r]);
            float fv = sigmoidf_(gf[r]);
            float gv = tanhf_(gg[r]);
            float ov = sigmoidf_(go[r]);
            float cn = fv * c[j][r] + iv * gv;
            c[j][r] = cn;
            mgTile[(r + 8 * p) * 64 + S * 32 + j * 16 + nn] =
                (unsigned short)f2bf1s(ov * tanhf_(cn));
        }
    }
    ldsfence();
}

// ---------- fused main kernel ----------
__global__ void __launch_bounds__(128) __attribute__((amdgpu_waves_per_eu(3)))
traj_disc_main(
    const float* __restrict__ traj, const float* __restrict__ state,
    const float* __restrict__ h0t, const float* __restrict__ c0t,
    const float* __restrict__ h0s, const float* __restrict__ c0s,
    const float* __restrict__ h0m, const float* __restrict__ c0m,
    const float* __restrict__ posW1, const float* __restrict__ posb1,
    const float* __restrict__ posW2, const float* __restrict__ posb2,
    const float* __restrict__ lightW1, const float* __restrict__ lightb1,
    const float* __restrict__ lightW2, const float* __restrict__ lightb2,
    const float* __restrict__ pWih, const float* __restrict__ pWhh,
    const float* __restrict__ pbih, const float* __restrict__ pbhh,
    const float* __restrict__ mWih, const float* __restrict__ mWhh,
    const float* __restrict__ mbih, const float* __restrict__ mbhh,
    const float* __restrict__ outW1, const float* __restrict__ outb1,
    const float* __restrict__ outW2, const float* __restrict__ outb2,
    const float* __restrict__ ssp, float* __restrict__ out) {
    extern __shared__ char smem[];
    unsigned* swz  = (unsigned*)smem;
    float*    rawf = (float*)(smem + RAWF_OFF);

    const int tid  = threadIdx.x;
    const int wave = tid >> 5;
    const int lane = tid & 31;

    // --- cooperative weight staging / swizzling ---
    for (int f = 0; f < 8; f++)  fillBfrag(swz + f * 256,        pWih, 16, 0, f, tid, 128);
    for (int f = 0; f < 8; f++)  fillBfrag(swz + (8 + f) * 256,  pWhh, 32, 0, f, tid, 128);
    for (int kt = 0; kt < 2; kt++)
        for (int nt = 0; nt < 16; nt++)
            fillBfrag(swz + (16 + kt * 16 + nt) * 256, mWih, 64, kt, nt, tid, 128);
    for (int kt = 0; kt < 2; kt++)
        for (int nt = 0; nt < 16; nt++)
            fillBfrag(swz + (48 + kt * 16 + nt) * 256, mWhh, 64, kt, nt, tid, 128);
    fillBfrag(swz + 80 * 256, posW2,   32, 0, 0, tid, 128);
    fillBfrag(swz + 81 * 256, lightW2, 32, 0, 0, tid, 128);
    for (int i = tid; i < 64;   i += 128) rawf[i] = posW1[i];
    for (int i = tid; i < 128;  i += 128) rawf[64 + i] = lightW1[i];
    if (tid < 32) rawf[192 + tid] = posb1[tid];
    if (tid < 32) rawf[224 + tid] = lightb1[tid];
    for (int i = tid; i < 2048; i += 128) rawf[256 + i] = outW1[i];
    if (tid < 32) rawf[2304 + tid] = outb1[tid];
    if (tid < 32) rawf[2336 + tid] = outW2[tid];
    if (tid == 0) rawf[2368] = outb2[0];

    // --- per-wave activation scratch ---
    char* wbase = smem + WSCR_OFF + wave * WSCR_SIZE;
    unsigned short* eTile  = (unsigned short*)wbase;            // [16][32] bf16 (K-padded)
    unsigned short* mgTile = (unsigned short*)(wbase + 1024);   // [16][64] bf16: h_traj|h_state
    unsigned short* hmTile = (unsigned short*)(wbase + 3072);   // [16][64] bf16: h_merge
    float*          xbn    = (float*)(wbase + 5120);            // [16][4]

    for (int i = lane; i < 16 * 32; i += 32) eTile[i] = 0;      // zero pad cols stay zero
    __syncthreads();

    const int row0 = blockIdx.x * 64 + wave * 16;
    const int p = lane >> 4, nn = lane & 15;

    // --- initial hidden / cell state ---
    for (int i = lane; i < 16 * 32; i += 32) {
        int m = i >> 5, k = i & 31;
        mgTile[m * 64 + k]      = (unsigned short)f2bf1s(h0t[(size_t)(row0 + m) * 32 + k]);
        mgTile[m * 64 + 32 + k] = (unsigned short)f2bf1s(h0s[(size_t)(row0 + m) * 32 + k]);
    }
    for (int i = lane; i < 16 * 64; i += 32) {
        int m = i >> 6, k = i & 63;
        hmTile[m * 64 + k] = (unsigned short)f2bf1s(h0m[(size_t)(row0 + m) * 64 + k]);
    }
    v8f cT[2], cS[2], cM[4];
#pragma unroll
    for (int j = 0; j < 2; j++)
#pragma unroll
        for (int r = 0; r < 8; r++) {
            int row = r + 8 * p, col = j * 16 + nn;
            cT[j][r] = c0t[(size_t)(row0 + row) * 32 + col];
            cS[j][r] = c0s[(size_t)(row0 + row) * 32 + col];
        }
#pragma unroll
    for (int j = 0; j < 4; j++)
#pragma unroll
        for (int r = 0; r < 8; r++)
            cM[j][r] = c0m[(size_t)(row0 + r + 8 * p) * 64 + j * 16 + nn];

    float pb[8], mb[16];
#pragma unroll
    for (int n = 0; n < 8; n++)  pb[n] = pbih[n * 16 + nn] + pbhh[n * 16 + nn];
#pragma unroll
    for (int n = 0; n < 16; n++) mb[n] = mbih[n * 16 + nn] + mbhh[n * 16 + nn];
    float b2pos = posb2[nn], b2light = lightb2[nn];
    ldsfence();

    // --- time loop ---
    for (int t = 0; t < TT; t++) {
        part_stream_step<0>(t, row0, lane, traj,  ssp,      rawf, swz, eTile, mgTile, xbn, b2pos,   pb, cT);
        part_stream_step<1>(t, row0, lane, state, ssp + 80, rawf, swz, eTile, mgTile, xbn, b2light, pb, cS);

        // merge LSTM: [16x64] input (h_traj|h_state) and [16x64] recurrent.
        // Gate column-groups processed one at a time (4 live accumulators).
        v16bf ain0 = loadAfrag(mgTile, 64, 0,  lane);
        v16bf ain1 = loadAfrag(mgTile, 64, 32, lane);
        v16bf ah0  = loadAfrag(hmTile, 64, 0,  lane);
        v16bf ah1  = loadAfrag(hmTile, 64, 32, lane);
#pragma unroll
        for (int j = 0; j < 4; j++) {
            v8f gi, gf, gg, go;
#pragma unroll
            for (int r = 0; r < 8; r++) {
                gi[r] = mb[j]; gf[r] = mb[4 + j]; gg[r] = mb[8 + j]; go[r] = mb[12 + j];
            }
            gi = wmma_bf16(ain0, loadBfrag(swz, 16 + j,      lane), gi);
            gi = wmma_bf16(ain1, loadBfrag(swz, 32 + j,      lane), gi);
            gi = wmma_bf16(ah0,  loadBfrag(swz, 48 + j,      lane), gi);
            gi = wmma_bf16(ah1,  loadBfrag(swz, 64 + j,      lane), gi);
            schedbar();
            gf = wmma_bf16(ain0, loadBfrag(swz, 16 + 4 + j,  lane), gf);
            gf = wmma_bf16(ain1, loadBfrag(swz, 32 + 4 + j,  lane), gf);
            gf = wmma_bf16(ah0,  loadBfrag(swz, 48 + 4 + j,  lane), gf);
            gf = wmma_bf16(ah1,  loadBfrag(swz, 64 + 4 + j,  lane), gf);
            schedbar();
            gg = wmma_bf16(ain0, loadBfrag(swz, 16 + 8 + j,  lane), gg);
            gg = wmma_bf16(ain1, loadBfrag(swz, 32 + 8 + j,  lane), gg);
            gg = wmma_bf16(ah0,  loadBfrag(swz, 48 + 8 + j,  lane), gg);
            gg = wmma_bf16(ah1,  loadBfrag(swz, 64 + 8 + j,  lane), gg);
            schedbar();
            go = wmma_bf16(ain0, loadBfrag(swz, 16 + 12 + j, lane), go);
            go = wmma_bf16(ain1, loadBfrag(swz, 32 + 12 + j, lane), go);
            go = wmma_bf16(ah0,  loadBfrag(swz, 48 + 12 + j, lane), go);
            go = wmma_bf16(ah1,  loadBfrag(swz, 64 + 12 + j, lane), go);
            schedbar();
#pragma unroll
            for (int r = 0; r < 8; r++) {
                float iv = sigmoidf_(gi[r]);
                float fv = sigmoidf_(gf[r]);
                float gv = tanhf_(gg[r]);
                float ov = sigmoidf_(go[r]);
                float cn = fv * cM[j][r] + iv * gv;
                cM[j][r] = cn;
                hmTile[(r + 8 * p) * 64 + j * 16 + nn] =
                    (unsigned short)f2bf1s(ov * tanhf_(cn));
            }
        }
        ldsfence();
    }

    // --- output head: (hM @ W1^T + b1) @ W2^T + b2, no activations ---
    if (lane < 16) {
        int m = lane;
        float o2 = rawf[2368];
        for (int j = 0; j < 32; j++) {
            float acc = rawf[2304 + j];
            const float* w = rawf + 256 + j * 64;
#pragma unroll 8
            for (int k = 0; k < 64; k++) acc += bf2f(hmTile[m * 64 + k]) * w[k];
            o2 += acc * rawf[2336 + j];
        }
        out[row0 + m] = o2;
    }
}

// ---------- host launcher ----------
extern "C" void kernel_launch(void* const* d_in, const int* in_sizes, int n_in,
                              void* d_out, int out_size, void* d_ws, size_t ws_size,
                              hipStream_t stream) {
    (void)in_sizes; (void)n_in; (void)out_size; (void)ws_size;
    const float* traj    = (const float*)d_in[0];
    const float* state   = (const float*)d_in[1];
    // d_in[2] seq_start_end: unused by the reference computation
    const float* h0t     = (const float*)d_in[3];
    const float* c0t     = (const float*)d_in[4];
    const float* h0s     = (const float*)d_in[5];
    const float* c0s     = (const float*)d_in[6];
    const float* h0m     = (const float*)d_in[7];
    const float* c0m     = (const float*)d_in[8];
    const float* pos_g   = (const float*)d_in[9];
    const float* pos_b   = (const float*)d_in[10];
    const float* posW1   = (const float*)d_in[11];
    const float* posb1   = (const float*)d_in[12];
    const float* posW2   = (const float*)d_in[13];
    const float* posb2   = (const float*)d_in[14];
    const float* light_g = (const float*)d_in[15];
    const float* light_b = (const float*)d_in[16];
    const float* lightW1 = (const float*)d_in[17];
    const float* lightb1 = (const float*)d_in[18];
    const float* lightW2 = (const float*)d_in[19];
    const float* lightb2 = (const float*)d_in[20];
    const float* pWih    = (const float*)d_in[21];
    const float* pWhh    = (const float*)d_in[22];
    const float* pbih    = (const float*)d_in[23];
    const float* pbhh    = (const float*)d_in[24];
    const float* mWih    = (const float*)d_in[25];
    const float* mWhh    = (const float*)d_in[26];
    const float* mbih    = (const float*)d_in[27];
    const float* mbhh    = (const float*)d_in[28];
    const float* outW1   = (const float*)d_in[29];
    const float* outb1   = (const float*)d_in[30];
    const float* outW2   = (const float*)d_in[31];
    const float* outb2   = (const float*)d_in[32];

    float* acc = (float*)d_ws;        // [0..240) sum/sumsq accumulators
    float* ssp = acc + 256;           // [0..240) scale/shift

    zero_kernel<<<2, 256, 0, stream>>>(acc);
    stats_kernel<<<TT * 16, 256, 0, stream>>>(traj, state, acc);
    finalize_kernel<<<1, 128, 0, stream>>>(acc, pos_g, pos_b, light_g, light_b, ssp);

    hipFuncSetAttribute(reinterpret_cast<const void*>(traj_disc_main),
                        hipFuncAttributeMaxDynamicSharedMemorySize, SMEM_BYTES);
    traj_disc_main<<<BB / 64, 128, SMEM_BYTES, stream>>>(
        traj, state, h0t, c0t, h0s, c0s, h0m, c0m,
        posW1, posb1, posW2, posb2, lightW1, lightb1, lightW2, lightb2,
        pWih, pWhh, pbih, pbhh, mWih, mWhh, mbih, mbhh,
        outW1, outb1, outW2, outb2, ssp, (float*)d_out);
}
